// DenseContrastLoss_3221225472055
// MI455X (gfx1250) — matrix operand order repaired and checked by reading, not compiled, not run on hardware
//
#include <hip/hip_runtime.h>
#include <hip/hip_bf16.h>

#define N_INST   100
#define CONV_DIM 256
#define SIDE     28
#define SS       (SIDE * SIDE)
#define NS       32
#define NSAMP    96
#define ROWS_PB  48            // sample rows per k_mlp block (2 blocks per instance)
#define TAU      0.07f
#define LOSS_W   1.2f
#define XS       264           // padded LDS row stride in ushorts (528 B, 16B-aligned)

#if defined(__has_builtin)
#if __has_builtin(__builtin_amdgcn_global_load_async_to_lds_b128)
#define HAVE_ASYNC_LDS 1
#endif
#endif
#ifndef HAVE_ASYNC_LDS
#define HAVE_ASYNC_LDS 0
#endif

typedef __bf16 bf16_t;
typedef bf16_t v16bf __attribute__((ext_vector_type(16)));
typedef float  v8f   __attribute__((ext_vector_type(8)));
typedef int    v4i   __attribute__((ext_vector_type(4)));

typedef __attribute__((address_space(1))) v4i* gbl_v4i_p;   // global
typedef __attribute__((address_space(3))) v4i* lds_v4i_p;   // LDS

union Frag {
    uint4 q[2];
    unsigned short h[16];
    v16bf v;
};

__device__ __forceinline__ unsigned short f32_to_bf16_bits(float f) {
    unsigned int u = __float_as_uint(f);
    u += 0x7FFFu + ((u >> 16) & 1u);   // round-to-nearest-even
    return (unsigned short)(u >> 16);
}

// ---------------- kernel 0: weights f32 -> bf16 ----------------
__global__ void k_cvt_weights(const float* __restrict__ w1, const float* __restrict__ w2,
                              unsigned short* __restrict__ w1b, unsigned short* __restrict__ w2b) {
    int i = blockIdx.x * blockDim.x + threadIdx.x;
    if (i < CONV_DIM * CONV_DIM) {
        w1b[i] = f32_to_bf16_bits(w1[i]);
        w2b[i] = f32_to_bf16_bits(w2[i]);
    }
}

// ---------------- kernel 1: gather sampled pixels -> packed bf16 X [N][96][256] ----------------
__global__ void k_gather(const float* __restrict__ feats,
                         const int* __restrict__ ai, const int* __restrict__ pi,
                         const int* __restrict__ ni, unsigned short* __restrict__ xb) {
    int n = blockIdx.x, tid = threadIdx.x;
    __shared__ int pix[NSAMP];
    if (tid < NSAMP) {
        int s = tid;
        const int* src = (s < NS) ? ai : ((s < 2 * NS) ? pi : ni);
        int so = s & (NS - 1);
        int y = src[(n * NS + so) * 2 + 0];
        int x = src[(n * NS + so) * 2 + 1];
        pix[s] = y * SIDE + x;
    }
    __syncthreads();
    const float* f = feats + (size_t)n * CONV_DIM * SS;
    unsigned short* xo = xb + (size_t)n * NSAMP * CONV_DIM;
    for (int idx = tid; idx < NSAMP * CONV_DIM; idx += blockDim.x) {
        int s = idx >> 8, c = idx & 255;
        xo[s * CONV_DIM + c] = f32_to_bf16_bits(f[c * SS + pix[s]]);
    }
}

// ---------------- kernel 2: fused 2-layer MLP via WMMA ----------------
// block = (instance, half): 48 sample rows. 256 threads = 8 waves.
// X staged to LDS via async global->LDS (ASYNCcnt path) when available.
__global__ void __launch_bounds__(256) k_mlp(const unsigned short* __restrict__ xb,
                                             const unsigned short* __restrict__ w1b,
                                             const float* __restrict__ b1,
                                             const unsigned short* __restrict__ w2b,
                                             const float* __restrict__ b2,
                                             float* __restrict__ pf) {
    int blk = blockIdx.x;
    int n = blk >> 1;
    int r0 = (blk & 1) * ROWS_PB;
    int tid = threadIdx.x;
    int lane = tid & 31, wave = tid >> 5;
    int ln = lane & 15, hi = lane >> 4;

    __shared__ unsigned short Xs[ROWS_PB * XS];   // 25344 B
    __shared__ unsigned short Hs[ROWS_PB * XS];   // 25344 B

    const unsigned short* Xg = xb + ((size_t)n * NSAMP + r0) * CONV_DIM;

    // ---- stage X rows into LDS (padded) ----
#if HAVE_ASYNC_LDS
    for (int r = wave; r < ROWS_PB; r += 8) {
        __builtin_amdgcn_global_load_async_to_lds_b128(
            (gbl_v4i_p)(void*)(Xg + r * CONV_DIM + lane * 8),    // 16 B per lane
            (lds_v4i_p)(void*)(Xs + r * XS + lane * 8),
            0, 0);
    }
    asm volatile("s_wait_asynccnt 0" ::: "memory");
#else
    for (int idx = tid; idx < ROWS_PB * (CONV_DIM / 8); idx += 256) {
        int r = idx >> 5, c8 = idx & 31;                         // 32 x 16B chunks/row
        *(uint4*)(Xs + r * XS + c8 * 8) = *(const uint4*)(Xg + r * CONV_DIM + c8 * 8);
    }
#endif
    __syncthreads();

    // ---- layer 1: H = relu(X @ W1^T + b1), H -> LDS ----
    for (int t = wave; t < 48; t += 8) {           // 3 Mt x 16 Nt tiles
        int mt = t >> 4, nt = t & 15;
        float bias = b1[nt * 16 + ln];
        v8f acc;
#pragma unroll
        for (int r = 0; r < 8; ++r) acc[r] = bias;
        const unsigned short* arow = Xs + (mt * 16 + ln) * XS;
        const unsigned short* brow = w1b + (nt * 16 + ln) * CONV_DIM;
#pragma unroll
        for (int kk = 0; kk < 8; ++kk) {
            int c0 = kk * 32;
            Frag a, b;
            a.q[0] = *(const uint4*)(arow + c0 + hi * 8);
            a.q[1] = *(const uint4*)(arow + c0 + 16 + hi * 8);
            b.q[0] = *(const uint4*)(brow + c0 + hi * 8);
            b.q[1] = *(const uint4*)(brow + c0 + 16 + hi * 8);
            acc = __builtin_amdgcn_wmma_f32_16x16x32_bf16(false, a.v, false, b.v,
                                                          (short)0, acc, false, false);
        }
#pragma unroll
        for (int r = 0; r < 8; ++r) {
            int m = mt * 16 + r + hi * 8;
            Hs[m * XS + nt * 16 + ln] = f32_to_bf16_bits(fmaxf(acc[r], 0.0f));
        }
    }
    __syncthreads();

    // ---- layer 2: P = H @ W2^T + b2 -> global f32 ----
    float* P = pf + ((size_t)n * NSAMP + r0) * CONV_DIM;
    for (int t = wave; t < 48; t += 8) {
        int mt = t >> 4, nt = t & 15;
        float bias = b2[nt * 16 + ln];
        v8f acc;
#pragma unroll
        for (int r = 0; r < 8; ++r) acc[r] = bias;
        const unsigned short* arow = Hs + (mt * 16 + ln) * XS;
        const unsigned short* brow = w2b + (nt * 16 + ln) * CONV_DIM;
#pragma unroll
        for (int kk = 0; kk < 8; ++kk) {
            int c0 = kk * 32;
            Frag a, b;
            a.q[0] = *(const uint4*)(arow + c0 + hi * 8);
            a.q[1] = *(const uint4*)(arow + c0 + 16 + hi * 8);
            b.q[0] = *(const uint4*)(brow + c0 + hi * 8);
            b.q[1] = *(const uint4*)(brow + c0 + 16 + hi * 8);
            acc = __builtin_amdgcn_wmma_f32_16x16x32_bf16(false, a.v, false, b.v,
                                                          (short)0, acc, false, false);
        }
#pragma unroll
        for (int r = 0; r < 8; ++r) {
            int m = mt * 16 + r + hi * 8;
            P[m * CONV_DIM + nt * 16 + ln] = acc[r];
        }
    }
}

// ---------------- kernel 3: norms + sims via WMMA + InfoNCE loss; block = 128 (4 waves) ----------------
__global__ void __launch_bounds__(128) k_loss(const float* __restrict__ pf,
                                              const unsigned char* __restrict__ gt,
                                              float* __restrict__ lp, float* __restrict__ vf) {
    int n = blockIdx.x, tid = threadIdx.x;
    int lane = tid & 31, wave = tid >> 5;
    int ln = lane & 15, hi = lane >> 4;
    __shared__ float s_ap[32 * 32], s_an[32 * 32], nloc[NSAMP], red[32];
    __shared__ int area;

    const float* P = pf + (size_t)n * NSAMP * CONV_DIM;

    if (tid == 0) area = 0;
    // per-row L2 norms (clamped at 1e-12), one row per thread
    if (tid < NSAMP) {
        const float* row = P + tid * CONV_DIM;
        float s = 0.f;
        for (int c = 0; c < CONV_DIM; ++c) { float v = row[c]; s += v * v; }
        nloc[tid] = fmaxf(sqrtf(s), 1e-12f);
    }
    __syncthreads();

    int pa = 0;
    for (int i = tid; i < SS; i += 128) pa += (int)gt[n * SS + i];
    atomicAdd(&area, pa);

    for (int pass = 0; pass < 2; ++pass) {    // 0: anchor.pos^T, 1: anchor.neg^T
        int mt = wave >> 1, nt = wave & 1;    // 2x2 tiles of 16, one per wave
        int brow0 = pass ? 64 : 32;
        v8f acc = {};
        const float* arow = P + (mt * 16 + ln) * CONV_DIM;            // anchors: rows 0..31
        const float* brow = P + (brow0 + nt * 16 + ln) * CONV_DIM;    // pos/neg rows
#pragma unroll
        for (int kk = 0; kk < 8; ++kk) {
            int c0 = kk * 32;
            Frag a, b;
#pragma unroll
            for (int j = 0; j < 8; ++j) {
                a.h[j]     = f32_to_bf16_bits(arow[c0 + hi * 8 + j]);
                a.h[8 + j] = f32_to_bf16_bits(arow[c0 + 16 + hi * 8 + j]);
            }
#pragma unroll
            for (int j = 0; j < 8; ++j) {
                b.h[j]     = f32_to_bf16_bits(brow[c0 + hi * 8 + j]);
                b.h[8 + j] = f32_to_bf16_bits(brow[c0 + 16 + hi * 8 + j]);
            }
            acc = __builtin_amdgcn_wmma_f32_16x16x32_bf16(false, a.v, false, b.v,
                                                          (short)0, acc, false, false);
        }
        float* dst = pass ? s_an : s_ap;
        float nb = nloc[brow0 + nt * 16 + ln];
#pragma unroll
        for (int r = 0; r < 8; ++r) {
            int k = mt * 16 + r + hi * 8;                 // anchor index
            dst[k * 32 + nt * 16 + ln] = acc[r] / (TAU * nloc[k] * nb);
        }
    }
    __syncthreads();

    if (tid < 32) {
        int k = tid;
        float mx = -1e30f;
        for (int m = 0; m < 32; ++m) mx = fmaxf(mx, s_ap[k * 32 + m]);
        float S = 0.f;
        for (int m = 0; m < 32; ++m) S += expf(s_an[k * 32 + m] - mx);
        float accl = 0.f;
        for (int m = 0; m < 32; ++m) {
            float d = s_ap[k * 32 + m] - mx;
            accl += logf(expf(d) + S) - d;    // -log(e^d/(e^d+S))
        }
        red[k] = accl;
    }
    __syncthreads();
    if (tid == 0) {
        float s = 0.f;
        for (int k = 0; k < 32; ++k) s += red[k];
        float lper = s / 1024.0f;
        int a = area;
        int valid = (a > NS) && (a < SS - NS);
        lp[n] = valid ? lper : 0.0f;
        vf[n] = valid ? 1.0f : 0.0f;
    }
}

// ---------------- kernel 4: final masked mean ----------------
__global__ void k_final(const float* __restrict__ lp, const float* __restrict__ vf,
                        float* __restrict__ out) {
    if (threadIdx.x == 0 && blockIdx.x == 0) {
        float s = 0.f, nv = 0.f;
        for (int i = 0; i < N_INST; ++i) { s += lp[i]; nv += vf[i]; }
        float loss = (nv > 0.f) ? (s / fmaxf(nv, 1.0f)) : 0.0f;
        out[0] = loss * LOSS_W;
    }
}

extern "C" void kernel_launch(void* const* d_in, const int* in_sizes, int n_in,
                              void* d_out, int out_size, void* d_ws, size_t ws_size,
                              hipStream_t stream) {
    (void)in_sizes; (void)n_in; (void)out_size; (void)ws_size;
    const float* feats        = (const float*)d_in[0];
    const float* w1           = (const float*)d_in[1];
    const float* b1           = (const float*)d_in[2];
    const float* w2           = (const float*)d_in[3];
    const float* b2           = (const float*)d_in[4];
    const unsigned char* gt   = (const unsigned char*)d_in[5];
    const int* ai             = (const int*)d_in[6];
    const int* pi             = (const int*)d_in[7];
    const int* ni             = (const int*)d_in[8];

    char* ws = (char*)d_ws;
    unsigned short* w1b = (unsigned short*)(ws + 0);          // 131072 B
    unsigned short* w2b = (unsigned short*)(ws + 131072);     // 131072 B
    unsigned short* xb  = (unsigned short*)(ws + 262144);     // 100*96*256*2 = 4915200 B
    float*          pf  = (float*)(ws + 5177344);             // 100*96*256*4 = 9830400 B
    float*          lp  = (float*)(ws + 15007744);            // 400 B
    float*          vfl = (float*)(ws + 15008144);            // 400 B

    k_cvt_weights<<<(CONV_DIM * CONV_DIM + 255) / 256, 256, 0, stream>>>(w1, w2, w1b, w2b);
    k_gather<<<N_INST, 256, 0, stream>>>(feats, ai, pi, ni, xb);
    k_mlp<<<2 * N_INST, 256, 0, stream>>>(xb, w1b, b1, w2b, b2, pf);
    k_loss<<<N_INST, 128, 0, stream>>>(pf, gt, lp, vfl);
    k_final<<<1, 64, 0, stream>>>(lp, vfl, (float*)d_out);
}
